// RoPEMultiheadAttention_22582938043040
// MI455X (gfx1250) — matrix operand order repaired and checked by reading, not compiled, not run on hardware
//
#include <hip/hip_runtime.h>

// ---------------------------------------------------------------------------
// RoPE multi-head attention for MI455X (gfx1250), bf16 WMMA pipeline.
// B=2, S=2048, D=1024, H=16, hd=64.
// ---------------------------------------------------------------------------

typedef __attribute__((ext_vector_type(16))) __bf16 v16bf;
typedef __attribute__((ext_vector_type(8)))  __bf16 v8bf;
typedef __attribute__((ext_vector_type(8)))  float  v8f;

constexpr int Bc  = 2;
constexpr int Sc  = 2048;
constexpr int Dc  = 1024;
constexpr int Hc  = 16;
constexpr int HDc = 64;
constexpr int Mc  = Bc * Sc;   // 4096 rows of X

#define DEV static __device__ __forceinline__

DEV __bf16 f2bf(float f) {
  unsigned u = __float_as_uint(f);
  u += 0x7fffu + ((u >> 16) & 1u);            // round-to-nearest-even
  unsigned short h = (unsigned short)(u >> 16);
  return __builtin_bit_cast(__bf16, h);
}

// xor-swizzle across the 32 lanes of a wave (single ds_swizzle, imm pattern)
template <int M>
DEV float swz_xor(float x) {
  return __int_as_float(
      __builtin_amdgcn_ds_swizzle(__float_as_int(x), (M << 10) | 0x1f));
}

// A fragment: 16x32 (MxK). ISA 7.12.2: lane l holds row m=l&15; K index =
// (v>>2)*16 + (l>>4)*8 + (v&3)*2 (+0,+1) for VGPR v.
DEV v16bf load_a_frag(const __bf16* base, int lda) {
  const int lane = threadIdx.x & 31;
  const int m = lane & 15, hh = lane >> 4;
  v16bf a;
#pragma unroll
  for (int v = 0; v < 8; ++v) {
    const int k = (v >> 2) * 16 + hh * 8 + (v & 3) * 2;
    a[2 * v]     = base[m * lda + k];
    a[2 * v + 1] = base[m * lda + k + 1];
  }
  return a;
}

// B fragment: 32x16 (KxN), source stored transposed as [n][k]. Lane l holds
// column n=l&15; K index = (v>>2)*16 + (l>>4)*8 + (v&3)*2 (+0,+1) for VGPR v.
DEV v16bf load_b_frag(const __bf16* baseT, int ldb) {
  const int lane = threadIdx.x & 31;
  const int n = lane & 15, hh = lane >> 4;
  v16bf b;
#pragma unroll
  for (int v = 0; v < 8; ++v) {
    const int k = (v >> 2) * 16 + hh * 8 + (v & 3) * 2;
    b[2 * v]     = baseT[n * ldb + k];
    b[2 * v + 1] = baseT[n * ldb + k + 1];
  }
  return b;
}

DEV v8f wmma_bf16(v16bf a, v16bf b, v8f c) {
  return __builtin_amdgcn_wmma_f32_16x16x32_bf16(
      /*neg_a=*/false, a, /*neg_b=*/false, b,
      /*c_mod=*/(short)0, c, /*reuse_a=*/false, /*reuse_b=*/false);
}

// ---------------------------------------------------------------------------
// fp32 -> bf16 conversion, 8 elements/thread (b128 in, b128 out)
// ---------------------------------------------------------------------------
__global__ void cvt_f32_bf16(const float* __restrict__ in,
                             __bf16* __restrict__ out, int n) {
  int i = (blockIdx.x * blockDim.x + threadIdx.x) * 8;
  const int stride = gridDim.x * blockDim.x * 8;
  for (; i < n; i += stride) {
    const float4 a = *(const float4*)(in + i);
    const float4 b = *(const float4*)(in + i + 4);
    v8bf o;
    o[0] = f2bf(a.x); o[1] = f2bf(a.y); o[2] = f2bf(a.z); o[3] = f2bf(a.w);
    o[4] = f2bf(b.x); o[5] = f2bf(b.y); o[6] = f2bf(b.z); o[7] = f2bf(b.w);
    *(v8bf*)(out + i) = o;
  }
}

// ---------------------------------------------------------------------------
// Projection GEMM: Y = X (MxK) * W^T (W is NxK row-major) + bias, then *scale.
// Block tile 128x128, 256 threads = 8 waves, wave tile 32x64 (2x4 WMMA).
// MODE 0: RoPE epilogue, bf16 out in [b,h,s,d]
// MODE 1: bf16 out in [b,h,s,d]
// MODE 2: fp32 out in [m,n] (final projection -> d_out)
// ---------------------------------------------------------------------------
template <int MODE>
__global__ __launch_bounds__(256) void proj_gemm(
    const __bf16* __restrict__ X, const __bf16* __restrict__ W,
    const float* __restrict__ bias, void* __restrict__ outp, float scale) {
  constexpr int K = Dc, N = Dc;
  __shared__ __attribute__((aligned(16))) __bf16 As[128][48];  // 96B rows
  __shared__ __attribute__((aligned(16))) __bf16 Bs[128][48];  // [n][k]

  const int t  = threadIdx.x;
  const int w  = t >> 5;
  const int wm = w >> 1, wn = w & 1;
  const int m0 = blockIdx.x * 128;
  const int n0 = blockIdx.y * 128;

  v8f acc[2][4] = {};

  for (int k0 = 0; k0 < K; k0 += 32) {
    const int r = t >> 1, c = (t & 1) * 16;
    {  // A tile 128x32
      const uint4* src = (const uint4*)(X + (size_t)(m0 + r) * K + k0 + c);
      *(uint4*)(&As[r][c])     = src[0];
      *(uint4*)(&As[r][c + 8]) = src[1];
    }
    {  // B tile: rows of W (columns of W^T) -> Bs[n][k]
      const uint4* src = (const uint4*)(W + (size_t)(n0 + r) * K + k0 + c);
      *(uint4*)(&Bs[r][c])     = src[0];
      *(uint4*)(&Bs[r][c + 8]) = src[1];
    }
    __syncthreads();

    const v16bf a0 = load_a_frag(&As[wm * 32][0], 48);
    const v16bf a1 = load_a_frag(&As[wm * 32 + 16][0], 48);
#pragma unroll
    for (int j = 0; j < 4; ++j) {
      const v16bf bj = load_b_frag(&Bs[wn * 64 + j * 16][0], 48);
      acc[0][j] = wmma_bf16(a0, bj, acc[0][j]);
      acc[1][j] = wmma_bf16(a1, bj, acc[1][j]);
    }
    __syncthreads();
  }

  const int lane = t & 31;
  const int ln = lane & 15, lh = lane >> 4;
#pragma unroll
  for (int i = 0; i < 2; ++i) {
#pragma unroll
    for (int j = 0; j < 4; ++j) {
      const int mb = m0 + wm * 32 + i * 16;
      const int nb = n0 + wn * 64 + j * 16;
      const int n  = nb + ln;
      const float bn = bias[n];
#pragma unroll
      for (int v = 0; v < 8; ++v) {
        const int m = mb + v + 8 * lh;   // C layout: row = v + 8*(lane>=16)
        float val   = acc[i][j][v] + bn;
        if constexpr (MODE == 2) {
          ((float*)outp)[(size_t)m * N + n] = val;
        } else {
          const int b = m >> 11, s = m & (Sc - 1);
          const int h = n >> 6, d = n & 63;
          if constexpr (MODE == 0) {
            // RoPE: (even,odd) columns sit in adjacent lanes of the C tile.
            const float partner = swz_xor<1>(val);
            const int   pi  = d >> 1;
            const float inv = __powf(10000.0f, -(float)(2 * pi) * (1.0f / 64.0f));
            const float ang = (float)s * inv;
            float sn, cs;
            __sincosf(ang, &sn, &cs);
            val = ((d & 1) == 0) ? (val * cs - partner * sn)
                                 : (partner * sn + val * cs);
          }
          val *= scale;
          ((__bf16*)outp)[(((size_t)b * Hc + h) * Sc + s) * HDc + d] = f2bf(val);
        }
      }
    }
  }
}

// ---------------------------------------------------------------------------
// Flash attention. Grid (B*H, S/128); 256 threads = 8 waves; each wave owns
// 16 query rows. Score GEMM is computed transposed (A = K-tile, B = Q) so the
// C layout holds ONE q-row per lane: the softmax reduction is register-local
// plus a single xor-16 swizzle. K is consumed in natural [key][d] layout;
// only V needs an LDS transpose. P round-trips through per-wave LDS to become
// an A fragment for P*V.
// ---------------------------------------------------------------------------
__global__ __launch_bounds__(256) void attn_fa(
    const __bf16* __restrict__ Q, const __bf16* __restrict__ Kk,
    const __bf16* __restrict__ V, __bf16* __restrict__ ctx) {
  constexpr int KT = 64;                                       // key tile
  __shared__ __attribute__((aligned(16))) __bf16 Ks[64][80];   // [key][d]
  __shared__ __attribute__((aligned(16))) __bf16 VsT[64][80];  // [d][key]
  __shared__ __attribute__((aligned(16))) __bf16 Ps[8][16][80];// [wave][q][key]
  __shared__ float Al[8][16];                                  // per-row alpha

  const int t = threadIdx.x;
  const int w = t >> 5, lane = t & 31;
  const int ln = lane & 15, lh = lane >> 4;
  const int bh = blockIdx.x;                 // 0..B*H-1
  const int b = bh >> 4, h = bh & 15;
  const int qblk = blockIdx.y;               // 0..S/128-1
  const size_t base = (size_t)bh * Sc * HDc;
  const __bf16* qp = Q + base;
  const __bf16* kp = Kk + base;
  const __bf16* vp = V + base;

  // Q as B fragments (natural [q][d] layout), already scaled by hd^-0.5.
  const int m0 = qblk * 128 + w * 16;
  const v16bf qf0 = load_b_frag(qp + (size_t)m0 * 64, 64);
  const v16bf qf1 = load_b_frag(qp + (size_t)m0 * 64 + 32, 64);

  float mrun = -3.0e38f, lrun = 0.0f;        // this lane's q-row = m0 + ln
  v8f o[4] = {};                             // o[j]: cols d = j*16+ln

  for (int kt = 0; kt < Sc / KT; ++kt) {
    __syncthreads();  // previous iteration done with Ks/VsT/Ps
    {
      const int r = t >> 2, c = (t & 3) * 16;
      const uint4* ksrc = (const uint4*)(kp + (size_t)(kt * KT + r) * 64 + c);
      *(uint4*)(&Ks[r][c])     = ksrc[0];
      *(uint4*)(&Ks[r][c + 8]) = ksrc[1];
      const uint4* vsrc = (const uint4*)(vp + (size_t)(kt * KT + r) * 64 + c);
      uint4 tv0 = vsrc[0], tv1 = vsrc[1];
      const __bf16* te0 = (const __bf16*)&tv0;
      const __bf16* te1 = (const __bf16*)&tv1;
#pragma unroll
      for (int i2 = 0; i2 < 8; ++i2) {
        VsT[c + i2][r]     = te0[i2];        // transpose V into [d][key]
        VsT[c + 8 + i2][r] = te1[i2];
      }
    }
    __syncthreads();

    // scores^T: C rows = keys, cols = q-rows (one per lane)
    v8f s[4];
#pragma unroll
    for (int j = 0; j < 4; ++j) {
      s[j] = {};
      v16bf ka = load_a_frag(&Ks[j * 16][0], 80);
      s[j] = wmma_bf16(ka, qf0, s[j]);
      ka = load_a_frag(&Ks[j * 16][32], 80);
      s[j] = wmma_bf16(ka, qf1, s[j]);
    }

    // online softmax: this lane holds 32 of the 64 keys for q-row m0+ln
    float mx = -3.0e38f;
#pragma unroll
    for (int j = 0; j < 4; ++j)
#pragma unroll
      for (int v = 0; v < 8; ++v) mx = fmaxf(mx, s[j][v]);
    mx = fmaxf(mx, swz_xor<16>(mx));         // combine the two half-waves
    const float mnew  = fmaxf(mrun, mx);
    const float alpha = __expf(mrun - mnew);
    float rs = 0.0f;
#pragma unroll
    for (int j = 0; j < 4; ++j)
#pragma unroll
      for (int v = 0; v < 8; ++v) {
        const float p = __expf(s[j][v] - mnew);
        rs += p;
        Ps[w][ln][j * 16 + v + 8 * lh] = f2bf(p);   // P as [q][key]
      }
    rs += swz_xor<16>(rs);
    lrun = lrun * alpha + rs;
    mrun = mnew;
    if (lh == 0) Al[w][ln] = alpha;
    __syncthreads();  // Ps/Al visible (also keeps waves in step)

    float av[8];
#pragma unroll
    for (int v = 0; v < 8; ++v) av[v] = Al[w][v + 8 * lh];
#pragma unroll
    for (int j = 0; j < 4; ++j)
#pragma unroll
      for (int v = 0; v < 8; ++v) o[j][v] *= av[v];

    const v16bf pa0 = load_a_frag(&Ps[w][0][0], 80);
    const v16bf pa1 = load_a_frag(&Ps[w][0][32], 80);
#pragma unroll
    for (int j = 0; j < 4; ++j) {
      v16bf vb = load_b_frag(&VsT[j * 16][0], 80);
      o[j] = wmma_bf16(pa0, vb, o[j]);
      vb = load_b_frag(&VsT[j * 16][32], 80);
      o[j] = wmma_bf16(pa1, vb, o[j]);
    }
  }

  // broadcast per-row 1/l, normalize, write context [b][s][h*64+d]
  if (lh == 0) Al[w][ln] = lrun;
  __syncthreads();
#pragma unroll
  for (int v = 0; v < 8; ++v) {
    const float rl = 1.0f / Al[w][v + 8 * lh];
    const int s = m0 + v + 8 * lh;
    const size_t row = ((size_t)b * Sc + s) * Dc + (size_t)h * HDc;
#pragma unroll
    for (int j = 0; j < 4; ++j)
      ctx[row + j * 16 + ln] = f2bf(o[j][v] * rl);
  }
}

// ---------------------------------------------------------------------------
extern "C" void kernel_launch(void* const* d_in, const int* in_sizes, int n_in,
                              void* d_out, int out_size, void* d_ws,
                              size_t ws_size, hipStream_t stream) {
  const float* query = (const float*)d_in[0];
  const float* key   = (const float*)d_in[1];
  const float* value = (const float*)d_in[2];
  const float* q_w = (const float*)d_in[3];
  const float* q_b = (const float*)d_in[4];
  const float* k_w = (const float*)d_in[5];
  const float* k_b = (const float*)d_in[6];
  const float* v_w = (const float*)d_in[7];
  const float* v_b = (const float*)d_in[8];
  const float* o_w = (const float*)d_in[9];
  const float* o_b = (const float*)d_in[10];

  __bf16* ws = (__bf16*)d_ws;
  size_t off = 0;
  auto alloc = [&](size_t n) { __bf16* p = ws + off; off += n; return p; };
  const size_t MD = (size_t)Mc * Dc, DD = (size_t)Dc * Dc;
  __bf16* Xq = alloc(MD);
  __bf16* Xk = alloc(MD);
  __bf16* Xv = alloc(MD);
  __bf16* Wq = alloc(DD);
  __bf16* Wk = alloc(DD);
  __bf16* Wv = alloc(DD);
  __bf16* Wo = alloc(DD);
  __bf16* Qp = alloc(MD);   // [b,h,s,d] bf16, RoPE'd, pre-scaled
  __bf16* Kp = alloc(MD);   // [b,h,s,d] bf16, RoPE'd
  __bf16* Vp = alloc(MD);   // [b,h,s,d] bf16
  __bf16* Cx = alloc(MD);   // context, [m,n] bf16

  // 1) fp32 -> bf16
  cvt_f32_bf16<<<512, 256, 0, stream>>>(query, Xq, (int)MD);
  cvt_f32_bf16<<<512, 256, 0, stream>>>(key,   Xk, (int)MD);
  cvt_f32_bf16<<<512, 256, 0, stream>>>(value, Xv, (int)MD);
  cvt_f32_bf16<<<256, 256, 0, stream>>>(q_w, Wq, (int)DD);
  cvt_f32_bf16<<<256, 256, 0, stream>>>(k_w, Wk, (int)DD);
  cvt_f32_bf16<<<256, 256, 0, stream>>>(v_w, Wv, (int)DD);
  cvt_f32_bf16<<<256, 256, 0, stream>>>(o_w, Wo, (int)DD);

  // 2) q/k/v projections (fused bias + RoPE + head-split + q pre-scale)
  const dim3 gridP(Mc / 128, Dc / 128);
  proj_gemm<0><<<gridP, 256, 0, stream>>>(Xq, Wq, q_b, (void*)Qp, 0.125f);
  proj_gemm<0><<<gridP, 256, 0, stream>>>(Xk, Wk, k_b, (void*)Kp, 1.0f);
  proj_gemm<1><<<gridP, 256, 0, stream>>>(Xv, Wv, v_b, (void*)Vp, 1.0f);

  // 3) flash attention
  attn_fa<<<dim3(Bc * Hc, Sc / 128), 256, 0, stream>>>(Qp, Kp, Vp, Cx);

  // 4) output projection -> fp32 d_out
  proj_gemm<2><<<gridP, 256, 0, stream>>>(Cx, Wo, o_b, d_out, 1.0f);
}
